// MoEXLayer_82884278878587
// MI455X (gfx1250) — matrix operand order repaired and checked by reading, not compiled, not run on hardware
//
#include <hip/hip_runtime.h>
#include <hip/hip_bf16.h>
#include <math.h>

// Problem constants (from reference): B=2,S=2048,D=1024,H=4096,E=8,K=2
#define BB   2
#define SS   2048
#define TOK  (BB * SS)   // 4096 tokens
#define DD   1024
#define HH   4096
#define EE   8
#define KK   2

typedef __attribute__((ext_vector_type(16))) __bf16        v16bf;
typedef __attribute__((ext_vector_type(8)))  float         v8f;
typedef __attribute__((ext_vector_type(4)))  unsigned int  v4u;

union Frag { v4u u[2]; v16bf v; };  // 32 bytes: one 16x16x32 bf16 A/B fragment per lane

struct KStep {                       // all fragments for one 32-deep k step
    Frag a0, a1;                     // A for expert-slot 0 / 1 (16 tok x 32 k)
    Frag b[4];                       // B for 4 h-subtiles (32 k x 16 h)
};

__device__ __forceinline__ unsigned short f2bf(float f) {
    unsigned int u = __float_as_uint(f);
    unsigned int r = u + 0x7FFFu + ((u >> 16) & 1u);   // round-to-nearest-even
    return (unsigned short)(r >> 16);
}

// ---------------------------------------------------------------------------
// Stage 1a: per-column mean / variance of W over H (unbiased=False)
// ---------------------------------------------------------------------------
__global__ void colstats_kernel(const float* __restrict__ W,
                                float* __restrict__ mW, float* __restrict__ vW) {
    __shared__ float smS[4][64];
    __shared__ float smQ[4][64];
    int tx = threadIdx.x;
    int dl = tx & 63;
    int hg = tx >> 6;                 // 0..3
    int d  = blockIdx.x * 64 + dl;
    float s = 0.f, q = 0.f;
    for (int h = hg; h < HH; h += 4) {
        float w = W[(size_t)h * DD + d];
        s += w; q += w * w;
    }
    smS[hg][dl] = s; smQ[hg][dl] = q;
    __syncthreads();
    if (hg == 0) {
        float S = smS[0][dl] + smS[1][dl] + smS[2][dl] + smS[3][dl];
        float Q = smQ[0][dl] + smQ[1][dl] + smQ[2][dl] + smQ[3][dl];
        float m = S * (1.0f / HH);
        mW[d] = m;
        vW[d] = Q * (1.0f / HH) - m * m;
    }
}

// ---------------------------------------------------------------------------
// Stage 1b: mean(bias)
// ---------------------------------------------------------------------------
__global__ void biasmean_kernel(const float* __restrict__ bias, float* __restrict__ mb) {
    __shared__ float sm[256];
    float s = 0.f;
    for (int h = threadIdx.x; h < HH; h += 256) s += bias[h];
    sm[threadIdx.x] = s;
    __syncthreads();
    for (int o = 128; o; o >>= 1) {
        if (threadIdx.x < (unsigned)o) sm[threadIdx.x] += sm[threadIdx.x + o];
        __syncthreads();
    }
    if (threadIdx.x == 0) *mb = sm[0] * (1.0f / HH);
}

// ---------------------------------------------------------------------------
// Stage 2: W (f32) -> bf16
// ---------------------------------------------------------------------------
__global__ void w_to_bf16_kernel(const float* __restrict__ W,
                                 unsigned short* __restrict__ Wbf, int n) {
    int i = blockIdx.x * blockDim.x + threadIdx.x;
    int stride = gridDim.x * blockDim.x;
    for (; i < n; i += stride) Wbf[i] = f2bf(W[i]);
}

// ---------------------------------------------------------------------------
// Stage 3: router. One wave32 per token; full f32 precision.
// ---------------------------------------------------------------------------
__global__ void router_kernel(const float* __restrict__ x,
                              const float* __restrict__ alpha,
                              const float* __restrict__ beta,
                              const float* __restrict__ mW,
                              const float* __restrict__ vW,
                              const float* __restrict__ meanBias,
                              int* __restrict__ rIdx, float* __restrict__ rWgt) {
    int lane  = threadIdx.x & 31;
    int token = blockIdx.x * 8 + (threadIdx.x >> 5);
    const float* xr = x + (size_t)token * DD;

    float mu[EE], s2[EE];
#pragma unroll
    for (int e = 0; e < EE; ++e) { mu[e] = 0.f; s2[e] = 0.f; }

    for (int d = lane; d < DD; d += 32) {
        float xv = xr[d];
        float x2 = xv * xv;
        float mwd = mW[d], vwd = vW[d];
#pragma unroll
        for (int e = 0; e < EE; ++e) {
            float a = alpha[e * DD + d];
            float b = beta [e * DD + d];
            mu[e] += xv * (mwd * a + b);
            s2[e] += x2 * (a * a * vwd);
        }
    }
#pragma unroll
    for (int e = 0; e < EE; ++e) {
        for (int off = 16; off; off >>= 1) {
            mu[e] += __shfl_xor(mu[e], off, 32);
            s2[e] += __shfl_xor(s2[e], off, 32);
        }
    }
    if (lane == 0) {
        float mb = *meanBias;
        float logits[EE];
#pragma unroll
        for (int e = 0; e < EE; ++e) {
            float sig = sqrtf(s2[e] + 1e-8f);
            logits[e] = erff((mu[e] + mb) / (1.41421356237309515f * sig));
        }
        // top-2, ties -> lowest index first (strict > keeps first max) like lax.top_k
        int i0 = 0; float v0 = logits[0];
#pragma unroll
        for (int e = 1; e < EE; ++e) if (logits[e] > v0) { v0 = logits[e]; i0 = e; }
        int i1 = -1; float v1 = -1e30f;
#pragma unroll
        for (int e = 0; e < EE; ++e) if (e != i0 && logits[e] > v1) { v1 = logits[e]; i1 = e; }
        float t  = __expf(v1 - v0);           // softmax over [v0, v1], v0 >= v1
        float w0 = 1.0f / (1.0f + t);
        float w1 = t * w0;
        rIdx[token * 2 + 0] = i0;  rIdx[token * 2 + 1] = i1;
        rWgt[token * 2 + 0] = w0;  rWgt[token * 2 + 1] = w1;
    }
}

// ---------------------------------------------------------------------------
// Stage 4: Xe[token][slot][d] = bf16( x[t,d] * alpha[e(t,slot),d] )
// ---------------------------------------------------------------------------
__global__ void build_xe_kernel(const float* __restrict__ x,
                                const float* __restrict__ alpha,
                                const int* __restrict__ rIdx,
                                unsigned short* __restrict__ Xe, int n) {
    int i = blockIdx.x * blockDim.x + threadIdx.x;
    int stride = gridDim.x * blockDim.x;
    for (; i < n; i += stride) {
        int d  = i & (DD - 1);
        int tk = i >> 10;            // token*2 + slot
        int t  = tk >> 1;
        int e  = rIdx[tk];
        Xe[i] = f2bf(x[(size_t)t * DD + d] * alpha[(size_t)e * DD + d]);
    }
}

// ---------------------------------------------------------------------------
// Stage 5: main GEMM + epilogue via v_wmma_f32_16x16x32_bf16.
// Wave tile: 16 tokens x 64 H, 2 slots -> 8 f32 accumulators.
// Block: 8 waves spanning 8 token tiles at a COMMON h-range, so the heavier
// B stream (8 loads/step/wave) is identical across waves -> WGP$/L0 hits.
// Software pipelined: k unrolled by 64 with ping-pong fragment buffers so the
// next step's 12 B128 loads overlap the current step's 8 WMMAs (avoids the
// s_wait_loadcnt 0 before every WMMA seen in the un-pipelined version).
// Fragment layouts per CDNA5 ISA 7.12.2 (wave32):
//   A (16x32 bf16): lane half 0 -> K {0..7,16..23}, half 1 -> K {8..15,24..31}; M = lane&15
//   B (32x16 bf16): lane half 0 -> K {0..7,16..23}, half 1 -> K {8..15,24..31}; N = lane&15
//   C (16x16 f32):  vgpr r, half h -> M = r + 8h, N = lane&15
// ---------------------------------------------------------------------------
__global__ void moe_gemm_kernel(const unsigned short* __restrict__ Xe,
                                const unsigned short* __restrict__ Wbf,
                                const float* __restrict__ bias,
                                const float* __restrict__ rWgt,
                                float* __restrict__ out) {
    const int lane = threadIdx.x & 31;
    const int wave = threadIdx.x >> 5;                // 0..7
    const int t0   = blockIdx.x * 128 + wave * 16;    // token tile (per wave)
    const int h0   = blockIdx.y * 64;                 // H tile (shared by block)

    const int half = lane >> 4;                       // 0 or 1
    const int l16  = lane & 15;

    // A pointers (per slot): row = token t0+l16; interleaved chunk offset 8*half
    const unsigned short* pA0 = Xe + (((size_t)(t0 + l16) * 2 + 0) * DD) + half * 8;
    const unsigned short* pA1 = Xe + (((size_t)(t0 + l16) * 2 + 1) * DD) + half * 8;
    // B pointers: column n = l16 -> W row h; interleaved chunk offset 8*half
    const unsigned short* pB[4];
#pragma unroll
    for (int j = 0; j < 4; ++j)
        pB[j] = Wbf + (size_t)(h0 + 16 * j + l16) * DD + half * 8;

    v8f acc[4][2] = {};

    auto loadStep = [&](int k0, KStep& f) {
        f.a0.u[0] = *(const v4u*)(pA0 + k0);
        f.a0.u[1] = *(const v4u*)(pA0 + k0 + 16);
        f.a1.u[0] = *(const v4u*)(pA1 + k0);
        f.a1.u[1] = *(const v4u*)(pA1 + k0 + 16);
#pragma unroll
        for (int j = 0; j < 4; ++j) {
            f.b[j].u[0] = *(const v4u*)(pB[j] + k0);
            f.b[j].u[1] = *(const v4u*)(pB[j] + k0 + 16);
        }
    };
    auto compute = [&](const KStep& f) {
#pragma unroll
        for (int j = 0; j < 4; ++j) {
            acc[j][0] = __builtin_amdgcn_wmma_f32_16x16x32_bf16(
                false, f.a0.v, false, f.b[j].v, (short)0, acc[j][0], false, false);
            acc[j][1] = __builtin_amdgcn_wmma_f32_16x16x32_bf16(
                false, f.a1.v, false, f.b[j].v, (short)0, acc[j][1], false, false);
        }
    };

    KStep f0, f1;
    loadStep(0, f0);
#pragma unroll 1
    for (int k0 = 0; k0 < DD - 64; k0 += 64) {
        loadStep(k0 + 32, f1);    // prefetch step n+1 while computing step n
        compute(f0);
        loadStep(k0 + 64, f0);    // prefetch step n+2 while computing step n+1
        compute(f1);
    }
    loadStep(DD - 32, f1);        // tail (k0 = DD-64 handled here)
    compute(f0);
    compute(f1);

    // Epilogue: out[t,h] = w0*relu(c0+bias) + w1*relu(c1+bias)
    float w0[8], w1[8];
#pragma unroll
    for (int r = 0; r < 8; ++r) {
        int t = t0 + r + half * 8;
        w0[r] = rWgt[t * 2 + 0];
        w1[r] = rWgt[t * 2 + 1];
    }
#pragma unroll
    for (int j = 0; j < 4; ++j) {
        int h = h0 + 16 * j + l16;
        float bv = bias[h];
#pragma unroll
        for (int r = 0; r < 8; ++r) {
            int t = t0 + r + half * 8;
            float p0 = acc[j][0][r] + bv; p0 = p0 > 0.f ? p0 : 0.f;
            float p1 = acc[j][1][r] + bv; p1 = p1 > 0.f ? p1 : 0.f;
            out[(size_t)t * HH + h] = w0[r] * p0 + w1[r] * p1;
        }
    }
}

// ---------------------------------------------------------------------------
extern "C" void kernel_launch(void* const* d_in, const int* in_sizes, int n_in,
                              void* d_out, int out_size, void* d_ws, size_t ws_size,
                              hipStream_t stream) {
    const float* x     = (const float*)d_in[0];   // [B,S,D]
    const float* W     = (const float*)d_in[1];   // [H,D]
    const float* bias  = (const float*)d_in[2];   // [H]
    const float* alpha = (const float*)d_in[3];   // [E,D]
    const float* beta  = (const float*)d_in[4];   // [E,D]
    float* out = (float*)d_out;                   // [B,S,H]

    // Workspace layout (~25.3 MB). d_ws is 256B-aligned from hipMalloc.
    char* ws = (char*)d_ws;
    float* mW       = (float*)ws;                 ws += DD * sizeof(float);
    float* vW       = (float*)ws;                 ws += DD * sizeof(float);
    float* meanBias = (float*)ws;                 ws += 4 * sizeof(float);
    int*   rIdx     = (int*)ws;                   ws += TOK * 2 * sizeof(int);
    float* rWgt     = (float*)ws;                 ws += TOK * 2 * sizeof(float);
    unsigned short* Wbf = (unsigned short*)ws;    ws += (size_t)HH * DD * sizeof(unsigned short);
    unsigned short* Xe  = (unsigned short*)ws;    // TOK*2*DD bf16

    colstats_kernel<<<DD / 64, 256, 0, stream>>>(W, mW, vW);
    biasmean_kernel<<<1, 256, 0, stream>>>(bias, meanBias);
    w_to_bf16_kernel<<<4096, 256, 0, stream>>>(W, Wbf, HH * DD);
    router_kernel<<<TOK / 8, 256, 0, stream>>>(x, alpha, beta, mW, vW, meanBias, rIdx, rWgt);
    build_xe_kernel<<<8192, 256, 0, stream>>>(x, alpha, rIdx, Xe, TOK * 2 * DD);
    moe_gemm_kernel<<<dim3(TOK / 128, HH / 64), 256, 0, stream>>>(Xe, Wbf, bias, rWgt, out);
}
